// RegionProposalNetwork_12893491823400
// MI455X (gfx1250) — matrix operand order repaired and checked by reading, not compiled, not run on hardware
//
#include <hip/hip_runtime.h>
#include <hip/hip_bf16.h>
#include <math.h>

// ---------------------------------------------------------------------------
// Problem constants (from reference setup_inputs)
// ---------------------------------------------------------------------------
constexpr int BN  = 16;            // batch
constexpr int CC  = 256;           // channels
constexpr int HH  = 100;           // feature H
constexpr int WW  = 152;           // feature W
constexpr int AA  = 15;            // anchors per cell
constexpr int HP  = HH + 2;        // padded H (halo)
constexpr int WP  = WW + 2;        // padded W (halo)
constexpr int HWN = HH * WW;       // 15200
constexpr int NSC = HWN * AA;      // 228000 scores per image
constexpr int KC  = 72;            // K chunks: 9 taps * 256 cin / 32
constexpr int PRE = 1000;          // top-k / post-NMS count
constexpr float DCLIP = 4.1351666f; // log(1000/16)

typedef __attribute__((ext_vector_type(16))) _Float16 v16h;
typedef __attribute__((ext_vector_type(8)))  _Float16 v8h;
typedef __attribute__((ext_vector_type(8)))  float    v8f;

// ---------------------------------------------------------------------------
// Workspace layout (bytes)
// ---------------------------------------------------------------------------
constexpr size_t alignUp(size_t x) { return (x + 255) & ~(size_t)255; }
constexpr size_t FT_OFF = 0;
constexpr size_t FT_SZ  = (size_t)BN * HP * WP * CC * 2;          // padded NHWC f16 features
constexpr size_t WF_OFF = alignUp(FT_OFF + FT_SZ);
constexpr size_t WF_SZ  = (size_t)KC * 16 * 32 * 16 * 2;          // fragment-packed f16 weights
constexpr size_t SC_OFF = alignUp(WF_OFF + WF_SZ);
constexpr size_t SC_SZ  = (size_t)BN * NSC * 4;                   // objectness scores
constexpr size_t DL_OFF = alignUp(SC_OFF + SC_SZ);
constexpr size_t DL_SZ  = (size_t)BN * NSC * 4 * 4;               // bbox deltas
constexpr size_t TS_OFF = alignUp(DL_OFF + DL_SZ);
constexpr size_t TS_SZ  = (size_t)BN * 1024 * 4;                  // top-k scores
constexpr size_t TI_OFF = alignUp(TS_OFF + TS_SZ);
constexpr size_t TI_SZ  = (size_t)BN * 1024 * 4;                  // top-k indices
constexpr size_t CT_OFF = alignUp(TI_OFF + TI_SZ);
constexpr size_t CT_SZ  = (size_t)BN * 8 * 4;                     // per-image counters
constexpr size_t PR_OFF = alignUp(CT_OFF + CT_SZ);
constexpr size_t PR_SZ  = (size_t)BN * PRE * 4 * 4;               // decoded proposals
constexpr size_t KP_OFF = alignUp(PR_OFF + PR_SZ);                // keep flags (BN*1024 ints)

// ---------------------------------------------------------------------------
// 1) NCHW f32 -> padded NHWC f16 features
// ---------------------------------------------------------------------------
__global__ void k_feat_f16(const float* __restrict__ f, _Float16* __restrict__ ft) {
  size_t idx = (size_t)blockIdx.x * blockDim.x + threadIdx.x;
  size_t total = (size_t)BN * HP * WP * CC;
  if (idx >= total) return;
  int c  = (int)(idx & (CC - 1));
  size_t r = idx >> 8;
  int xp = (int)(r % WP); r /= WP;
  int yp = (int)(r % HP);
  int b  = (int)(r / HP);
  float v = 0.f;
  int x = xp - 1, y = yp - 1;
  if (x >= 0 && x < WW && y >= 0 && y < HH)
    v = f[(((size_t)b * CC + c) * HH + y) * WW + x];
  ft[idx] = (_Float16)v;
}

// ---------------------------------------------------------------------------
// 2) Pack conv weights into WMMA B-fragment layout.
//    B 32x16 f16 fragment (wave32): lanes 0-15 hold N=lane, K=0..7,16..23 (2/VGPR);
//    lanes 16-31 hold N=lane-16, K=8..15,24..31.  Each lane's 16 halves are stored
//    contiguously: wf[((kc*16+nsub)*32+lane)*16 + j].
// ---------------------------------------------------------------------------
__global__ void k_wpack(const float* __restrict__ w, _Float16* __restrict__ wf) {
  int idx = blockIdx.x * blockDim.x + threadIdx.x;
  if (idx >= KC * 16 * 32 * 16) return;
  int j    = idx & 15;
  int lane = (idx >> 4) & 31;
  int nsub = (idx >> 9) & 15;
  int kc   = idx >> 13;
  int kk = (j < 8) ? (lane >> 4) * 8 + j : 16 + (lane >> 4) * 8 + (j - 8);         // K within chunk
  int co = nsub * 16 + (lane & 15);      // output channel
  int t  = kc >> 3;                      // tap 0..8 (ky*3+kx)
  int ci = (kc & 7) * 32 + kk;           // input channel
  wf[idx] = (_Float16)w[(size_t)co * (CC * 9) + (size_t)ci * 9 + t];
}

// ---------------------------------------------------------------------------
// 3) Implicit-GEMM 3x3 conv via v_wmma_f32_16x16x32_f16, fused ReLU + heads.
//    Block = 32 positions x 256 cout; 8 waves, each wave owns a 32x32 slab of
//    C (2 M-subtiles x 2 N-subtiles = 4 accumulators) so each B fragment is
//    reused by two A fragments (4 WMMAs per weight fetch).
//    K loop: nested (ty, tx, c) -- no divisions, inner 8 chunks unrolled.
// ---------------------------------------------------------------------------
__global__ __launch_bounds__(256)
void k_conv_wmma(const _Float16* __restrict__ ft, const _Float16* __restrict__ wf,
                 const float* __restrict__ cb,
                 const float* __restrict__ cw, const float* __restrict__ clb,
                 const float* __restrict__ bw, const float* __restrict__ bbb,
                 float* __restrict__ sc, float* __restrict__ dl) {
  __shared__ float tl[32][CC + 8];
  const int tid  = threadIdx.x;
  const int wave = tid >> 5;
  const int lane = tid & 31;
  const int b  = blockIdx.z;
  const int y  = blockIdx.y;
  const int x0 = blockIdx.x * 32;
  const int m  = lane & 15;        // A-fragment row within M-subtile
  const int kh = lane >> 4;        // A-fragment K-half selector
  const int n0 = wave * 32;        // this wave's output-channel base

  v8f acc00 = {};  // M-subtile 0, N-subtile 0
  v8f acc01 = {};  // M-subtile 0, N-subtile 1
  v8f acc10 = {};  // M-subtile 1, N-subtile 0
  v8f acc11 = {};  // M-subtile 1, N-subtile 1
  const size_t rowStride = (size_t)WP * CC;
  const _Float16* fbase = ft + (size_t)b * HP * rowStride;
  const _Float16* wbase = wf + ((size_t)(n0 >> 4) * 32 + lane) * 16;

  int kc = 0;
#pragma unroll 1
  for (int ty = 0; ty < 3; ++ty) {
    const _Float16* rp = fbase + (size_t)(y + ty) * rowStride;
#pragma unroll 1
    for (int tx = 0; tx < 3; ++tx) {
      int xa0 = x0 + m + tx;       if (xa0 > WP - 1) xa0 = WP - 1;
      int xa1 = x0 + 16 + m + tx;  if (xa1 > WP - 1) xa1 = WP - 1;
      const _Float16* ap0 = rp + (size_t)xa0 * CC + kh * 8;
      const _Float16* ap1 = rp + (size_t)xa1 * CC + kh * 8;
#pragma unroll
      for (int c = 0; c < 8; ++c, ++kc) {
        const int cc = c * 32;
        // A fragments: lane holds K = {kh*8..kh*8+7, 16+kh*8..16+kh*8+7}
        v8h a0lo = *(const v8h*)(ap0 + cc);
        v8h a0hi = *(const v8h*)(ap0 + cc + 16);
        v8h a1lo = *(const v8h*)(ap1 + cc);
        v8h a1hi = *(const v8h*)(ap1 + cc + 16);
        v16h a0, a1;
#pragma unroll
        for (int i = 0; i < 8; ++i) {
          a0[i] = a0lo[i]; a0[8 + i] = a0hi[i];
          a1[i] = a1lo[i]; a1[8 + i] = a1hi[i];
        }
        // B fragments: 32 contiguous bytes per lane, coalesced, L2-resident
        const _Float16* bp = wbase + (size_t)kc * (16 * 32 * 16);
        v16h b0 = *(const v16h*)bp;
        v16h b1 = *(const v16h*)(bp + 32 * 16);
        acc00 = __builtin_amdgcn_wmma_f32_16x16x32_f16(false, a0, false, b0, (short)0, acc00, false, false);
        acc01 = __builtin_amdgcn_wmma_f32_16x16x32_f16(false, a0, false, b1, (short)0, acc01, false, false);
        acc10 = __builtin_amdgcn_wmma_f32_16x16x32_f16(false, a1, false, b0, (short)0, acc10, false, false);
        acc11 = __builtin_amdgcn_wmma_f32_16x16x32_f16(false, a1, false, b1, (short)0, acc11, false, false);
      }
    }
  }

  // C/D layout: lanes 0-15: VGPR v -> M=v, N=lane; lanes 16-31: M=8+v, N=lane-16
  {
    const int nn = n0 + (lane & 15);
#pragma unroll
    for (int v = 0; v < 8; ++v) {
      const int mo = kh * 8 + v;
      float z;
      z = acc00[v] + cb[nn];      tl[mo][nn]           = z > 0.f ? z : 0.f;
      z = acc01[v] + cb[nn + 16]; tl[mo][nn + 16]      = z > 0.f ? z : 0.f;
      z = acc10[v] + cb[nn];      tl[16 + mo][nn]      = z > 0.f ? z : 0.f;
      z = acc11[v] + cb[nn + 16]; tl[16 + mo][nn + 16] = z > 0.f ? z : 0.f;
    }
  }
  __syncthreads();

  // Fused 1x1 heads: 15 cls logits + 60 bbox deltas per position.
  for (int u = tid; u < 32 * 80; u += 256) {
    const int pos = u / 80;
    const int o   = u - pos * 80;
    const int x   = x0 + pos;
    if (x >= WW || o >= 75) continue;
    const float* wrow;
    float bias;
    if (o < 15) { wrow = cw + (size_t)o * CC;        bias = clb[o]; }
    else        { wrow = bw + (size_t)(o - 15) * CC; bias = bbb[o - 15]; }
    float s = bias;
#pragma unroll 8
    for (int k = 0; k < CC; ++k) s += tl[pos][k] * wrow[k];
    const size_t p = (size_t)y * WW + x;
    if (o < 15) sc[(size_t)b * NSC + p * AA + o] = s;                 // obj[b, p*15+a]
    else        dl[(size_t)b * NSC * 4 + p * 60 + (o - 15)] = s;      // dl[b, p*15+a, d]
  }
}

// ---------------------------------------------------------------------------
// 4) Top-k pass 1: per-image 1024-bin histogram on sortable float keys.
// ---------------------------------------------------------------------------
__device__ __forceinline__ unsigned fkey(float f) {
  unsigned u = __float_as_uint(f);
  return (u & 0x80000000u) ? ~u : (u | 0x80000000u);
}

__global__ void k_topk_hist(const float* __restrict__ sc, int* __restrict__ ctr) {
  __shared__ int hist[1024];
  const int b = blockIdx.x, tid = threadIdx.x;
  for (int i = tid; i < 1024; i += 256) hist[i] = 0;
  __syncthreads();
  const float* s = sc + (size_t)b * NSC;
  for (int i = tid; i < NSC; i += 256) atomicAdd(&hist[fkey(s[i]) >> 22], 1);
  __syncthreads();
  if (tid == 0) {
    int acc = 0, T = 0;
    for (int bin = 1023; bin >= 0; --bin) {
      if (acc + hist[bin] >= PRE) { T = bin; break; }
      acc += hist[bin];
    }
    int* c = ctr + b * 8;
    c[0] = T; c[1] = acc; c[2] = PRE - acc; c[3] = 0; c[4] = 0;
  }
}

// ---------------------------------------------------------------------------
// 5) Top-k pass 2: compact exactly PRE candidates per image.
// ---------------------------------------------------------------------------
__global__ void k_topk_select(const float* __restrict__ sc, int* __restrict__ ctr,
                              float* __restrict__ ts, int* __restrict__ ti) {
  const int b = blockIdx.y;
  const int i = blockIdx.x * blockDim.x + threadIdx.x;
  if (i >= NSC) return;
  int* c = ctr + b * 8;
  const int T = c[0], hiN = c[1], quota = c[2];
  const float v = sc[(size_t)b * NSC + i];
  const int bin = (int)(fkey(v) >> 22);
  int slot = -1;
  if (bin > T) {
    slot = atomicAdd(c + 3, 1);
  } else if (bin == T) {
    int e = atomicAdd(c + 4, 1);
    if (e < quota) slot = hiN + e;
  }
  if (slot >= 0 && slot < PRE) { ts[b * 1024 + slot] = v; ti[b * 1024 + slot] = i; }
}

// ---------------------------------------------------------------------------
// 6) Top-k pass 3: bitonic sort 1024 (padded) descending by score.
// ---------------------------------------------------------------------------
__global__ void k_topk_sort(float* __restrict__ ts, int* __restrict__ ti) {
  __shared__ float s[1024];
  __shared__ int  id[1024];
  const int b = blockIdx.x, tid = threadIdx.x;
  for (int i = tid; i < 1024; i += 256) {
    if (i < PRE) { s[i] = ts[b * 1024 + i]; id[i] = ti[b * 1024 + i]; }
    else         { s[i] = -3.4e38f;         id[i] = 0; }
  }
  for (int k = 2; k <= 1024; k <<= 1) {
    for (int j = k >> 1; j > 0; j >>= 1) {
      __syncthreads();
      for (int i = tid; i < 1024; i += 256) {
        const int ixj = i ^ j;
        if (ixj > i) {
          const bool desc = ((i & k) == 0);
          const float a = s[i], c = s[ixj];
          if ((a < c) == desc) {
            s[i] = c; s[ixj] = a;
            int t2 = id[i]; id[i] = id[ixj]; id[ixj] = t2;
          }
        }
      }
    }
  }
  __syncthreads();
  for (int i = tid; i < PRE; i += 256) { ts[b * 1024 + i] = s[i]; ti[b * 1024 + i] = id[i]; }
}

// ---------------------------------------------------------------------------
// 7) Decode top proposals: analytic anchors + delta decode + clip + validity.
// ---------------------------------------------------------------------------
__global__ void k_decode(const float* __restrict__ dl, const int* __restrict__ ti,
                         const int* __restrict__ imh, const int* __restrict__ imw,
                         float* __restrict__ props, int* __restrict__ keep) {
  const int b = blockIdx.x, tid = threadIdx.x;
  const float IH = (float)imh[0], IW = (float)imw[0];
  const int sh = imh[0] / HH, sw = imw[0] / WW;
  const float SZ[5] = {32.f, 64.f, 128.f, 256.f, 512.f};
  const float RT[3] = {0.5f, 1.f, 2.f};
  for (int j = tid; j < PRE; j += 256) {
    const int idx = ti[b * 1024 + j];
    const int p = idx / AA, a = idx - p * AA;
    const int yq = p / WW, xq = p - yq * WW;
    const int r = a / 5, sI = a - r * 5;
    const float hr  = sqrtf(RT[r]);
    const float wsz = SZ[sI] / hr, hsz = SZ[sI] * hr;
    const float b0 = rintf(-wsz * 0.5f), b1 = rintf(-hsz * 0.5f);
    const float b2 = rintf( wsz * 0.5f), b3 = rintf( hsz * 0.5f);
    const float aw = b2 - b0, ah = b3 - b1;
    const float ax = (float)(xq * sw) + b0 + 0.5f * aw;
    const float ay = (float)(yq * sh) + b1 + 0.5f * ah;
    const float* d = dl + (size_t)b * NSC * 4 + (size_t)idx * 4;
    const float dx = d[0] / 10.f, dy = d[1] / 10.f;
    const float dw = fminf(d[2] / 5.f, DCLIP), dh = fminf(d[3] / 5.f, DCLIP);
    const float px = dx * aw + ax, py = dy * ah + ay;
    const float pw = expf(dw) * aw, ph = expf(dh) * ah;
    float x1 = px - 0.5f * pw, y1 = py - 0.5f * ph;
    float x2 = px + 0.5f * pw, y2 = py + 0.5f * ph;
    x1 = fminf(fmaxf(x1, 0.f), IW); x2 = fminf(fmaxf(x2, 0.f), IW);
    y1 = fminf(fmaxf(y1, 0.f), IH); y2 = fminf(fmaxf(y2, 0.f), IH);
    float* pr = props + ((size_t)b * PRE + j) * 4;
    pr[0] = x1; pr[1] = y1; pr[2] = x2; pr[3] = y2;
    keep[b * 1024 + j] = ((x2 - x1) >= 1e-3f && (y2 - y1) >= 1e-3f) ? 1 : 0;
  }
}

// ---------------------------------------------------------------------------
// 8) Greedy NMS: one 1024-thread block per image, boxes in LDS.
// ---------------------------------------------------------------------------
__global__ __launch_bounds__(1024)
void k_nms(const float* __restrict__ props, int* __restrict__ keep) {
  __shared__ float X1[PRE], Y1[PRE], X2[PRE], Y2[PRE], AR[PRE];
  __shared__ int   K[PRE];
  const int b = blockIdx.x, tid = threadIdx.x;
  for (int j = tid; j < PRE; j += 1024) {
    const float* pr = props + ((size_t)b * PRE + j) * 4;
    X1[j] = pr[0]; Y1[j] = pr[1]; X2[j] = pr[2]; Y2[j] = pr[3];
    AR[j] = (pr[2] - pr[0]) * (pr[3] - pr[1]);
    K[j]  = keep[b * 1024 + j];
  }
  for (int i = 0; i < PRE; ++i) {
    __syncthreads();
    if (!K[i]) continue;          // uniform (LDS) across the block
    const int j = tid;
    if (j > i && j < PRE) {
      float iw = fminf(X2[i], X2[j]) - fmaxf(X1[i], X1[j]); iw = iw > 0.f ? iw : 0.f;
      float ih = fminf(Y2[i], Y2[j]) - fmaxf(Y1[i], Y1[j]); ih = ih > 0.f ? ih : 0.f;
      const float inter = iw * ih;
      const float iou = inter / (AR[i] + AR[j] - inter + 1e-9f);
      if (iou > 0.7f) K[j] = 0;
    }
  }
  __syncthreads();
  for (int j = tid; j < PRE; j += 1024) keep[b * 1024 + j] = K[j];
}

// ---------------------------------------------------------------------------
// 9) Stable kept-first reorder + masked output.
//    d_out = [boxes 16*1000*4 | scores 16*1000 | valid 16*1000] floats.
// ---------------------------------------------------------------------------
__global__ void k_final(const float* __restrict__ props, const float* __restrict__ ts,
                        const int* __restrict__ keep, float* __restrict__ out) {
  __shared__ int rank[PRE];
  const int b = blockIdx.x, tid = threadIdx.x;
  if (tid == 0) {
    int r = 0;
    for (int j = 0; j < PRE; ++j) { rank[j] = r; r += (keep[b * 1024 + j] != 0); }
  }
  __syncthreads();
  float* boxes  = out;
  float* scores = out + (size_t)BN * PRE * 4;
  float* valid  = out + (size_t)BN * PRE * 5;
  for (int j = tid; j < PRE; j += 256) {
    const size_t base = ((size_t)b * PRE + j) * 4;
    boxes[base] = 0.f; boxes[base + 1] = 0.f; boxes[base + 2] = 0.f; boxes[base + 3] = 0.f;
    scores[(size_t)b * PRE + j] = 0.f;
    valid [(size_t)b * PRE + j] = 0.f;
  }
  __syncthreads();
  for (int j = tid; j < PRE; j += 256) {
    if (keep[b * 1024 + j]) {
      const int d = rank[j];
      const size_t sbase = ((size_t)b * PRE + j) * 4;
      const size_t dbase = ((size_t)b * PRE + d) * 4;
      boxes[dbase]     = props[sbase];
      boxes[dbase + 1] = props[sbase + 1];
      boxes[dbase + 2] = props[sbase + 2];
      boxes[dbase + 3] = props[sbase + 3];
      scores[(size_t)b * PRE + d] = ts[b * 1024 + j];
      valid [(size_t)b * PRE + d] = 1.f;
    }
  }
}

// ---------------------------------------------------------------------------
// Host launcher
// ---------------------------------------------------------------------------
extern "C" void kernel_launch(void* const* d_in, const int* in_sizes, int n_in,
                              void* d_out, int out_size, void* d_ws, size_t ws_size,
                              hipStream_t stream) {
  const float* features = (const float*)d_in[0];
  const float* conv_w   = (const float*)d_in[1];
  const float* conv_b   = (const float*)d_in[2];
  const float* cls_w    = (const float*)d_in[3];
  const float* cls_b    = (const float*)d_in[4];
  const float* bbox_w   = (const float*)d_in[5];
  const float* bbox_b   = (const float*)d_in[6];
  const int*   image_h  = (const int*)d_in[7];
  const int*   image_w  = (const int*)d_in[8];
  float* out = (float*)d_out;

  char* ws = (char*)d_ws;
  _Float16* ft = (_Float16*)(ws + FT_OFF);
  _Float16* wf = (_Float16*)(ws + WF_OFF);
  float*    sc = (float*)(ws + SC_OFF);
  float*    dl = (float*)(ws + DL_OFF);
  float*    ts = (float*)(ws + TS_OFF);
  int*      ti = (int*)(ws + TI_OFF);
  int*      ct = (int*)(ws + CT_OFF);
  float*    pr = (float*)(ws + PR_OFF);
  int*      kp = (int*)(ws + KP_OFF);

  // 1) features -> padded NHWC f16
  {
    size_t total = (size_t)BN * HP * WP * CC;
    k_feat_f16<<<dim3((unsigned)((total + 255) / 256)), dim3(256), 0, stream>>>(features, ft);
  }
  // 2) weight repack
  k_wpack<<<dim3(KC * 16 * 32 * 16 / 256), dim3(256), 0, stream>>>(conv_w, wf);
  // 3) WMMA conv + fused heads (32-position x 256-channel tiles)
  k_conv_wmma<<<dim3((WW + 31) / 32, HH, BN), dim3(256), 0, stream>>>(
      ft, wf, conv_b, cls_w, cls_b, bbox_w, bbox_b, sc, dl);
  // 4-6) top-k
  k_topk_hist<<<dim3(BN), dim3(256), 0, stream>>>(sc, ct);
  k_topk_select<<<dim3((NSC + 255) / 256, BN), dim3(256), 0, stream>>>(sc, ct, ts, ti);
  k_topk_sort<<<dim3(BN), dim3(256), 0, stream>>>(ts, ti);
  // 7) decode
  k_decode<<<dim3(BN), dim3(256), 0, stream>>>(dl, ti, image_h, image_w, pr, kp);
  // 8) NMS
  k_nms<<<dim3(BN), dim3(1024), 0, stream>>>(pr, kp);
  // 9) finalize
  k_final<<<dim3(BN), dim3(256), 0, stream>>>(pr, ts, kp, out);

  (void)in_sizes; (void)n_in; (void)out_size; (void)ws_size;
}
